// SemanticNoveltyTracker_70205535420793
// MI455X (gfx1250) — matrix operand order repaired and checked by reading, not compiled, not run on hardware
//
#include <hip/hip_runtime.h>
#include <math.h>

// Problem constants (match reference)
#define Bn 8192
#define Dn 1024
#define Cn 4096

#define M_BLK 32          // embedding rows per block
#define LDSS  1032        // padded LDS row stride (elements); 1032*2B = 2064B -> conflict-free

typedef __attribute__((ext_vector_type(16))) __bf16 v16bf;
typedef __attribute__((ext_vector_type(8)))  float  v8f;

union FragU {
    uint4 q[2];
    v16bf v;
};

__device__ __forceinline__ unsigned short f2bf(float x) {
    // round-to-nearest-even fp32 -> bf16 (inputs are finite)
    unsigned u = __float_as_uint(x);
    u = (u + 0x7FFFu + ((u >> 16) & 1u)) >> 16;
    return (unsigned short)u;
}
__device__ __forceinline__ float bf2f(unsigned short h) {
    return __uint_as_float(((unsigned)h) << 16);
}

// -------------------------------------------------------------------------
// Kernel 1: per-row L2 normalize + split into bf16 (hi, lo) pairs.
// blockIdx.x in [0, Bn) -> embedding rows; [Bn, Bn+Cn) -> centroid rows.
// 256 threads, each handles 4 consecutive floats of the 1024-length row.
// -------------------------------------------------------------------------
__global__ __launch_bounds__(256) void norm_split_kernel(
    const float* __restrict__ emb, const float* __restrict__ cen,
    unsigned short* __restrict__ ehi, unsigned short* __restrict__ elo,
    unsigned short* __restrict__ chi, unsigned short* __restrict__ clo) {

    const int row = blockIdx.x;
    const float* src;
    unsigned short *dhi, *dlo;
    if (row < Bn) {
        src = emb + (size_t)row * Dn;
        dhi = ehi + (size_t)row * Dn;
        dlo = elo + (size_t)row * Dn;
    } else {
        const int r = row - Bn;
        src = cen + (size_t)r * Dn;
        dhi = chi + (size_t)r * Dn;
        dlo = clo + (size_t)r * Dn;
    }

    const int tid = threadIdx.x;
    const float4 x = ((const float4*)src)[tid];
    float s = x.x * x.x + x.y * x.y + x.z * x.z + x.w * x.w;

    // wave32 reduce
    #pragma unroll
    for (int m = 16; m >= 1; m >>= 1) s += __shfl_xor(s, m, 32);

    __shared__ float wsum[8];
    __shared__ float sscale;
    const int wid = tid >> 5, lid = tid & 31;
    if (lid == 0) wsum[wid] = s;
    __syncthreads();
    if (tid == 0) {
        float t = 0.0f;
        #pragma unroll
        for (int i = 0; i < 8; ++i) t += wsum[i];
        sscale = 1.0f / fmaxf(sqrtf(t), 1e-12f);   // matches F.normalize eps
    }
    __syncthreads();
    const float sc = sscale;

    const float y0 = x.x * sc, y1 = x.y * sc, y2 = x.z * sc, y3 = x.w * sc;
    const unsigned short h0 = f2bf(y0), h1 = f2bf(y1), h2 = f2bf(y2), h3 = f2bf(y3);
    const unsigned short l0 = f2bf(y0 - bf2f(h0));
    const unsigned short l1 = f2bf(y1 - bf2f(h1));
    const unsigned short l2 = f2bf(y2 - bf2f(h2));
    const unsigned short l3 = f2bf(y3 - bf2f(h3));

    uint2 ph, pl;
    ph.x = (unsigned)h0 | ((unsigned)h1 << 16);
    ph.y = (unsigned)h2 | ((unsigned)h3 << 16);
    pl.x = (unsigned)l0 | ((unsigned)l1 << 16);
    pl.y = (unsigned)l2 | ((unsigned)l3 << 16);
    ((uint2*)dhi)[tid] = ph;
    ((uint2*)dlo)[tid] = pl;
}

#define WMMA3(ACC, A_, AL_, BH_, BL_)                                               \
    ACC = __builtin_amdgcn_wmma_f32_16x16x32_bf16(false, (A_), false, (BH_),        \
                                                  (short)0, ACC, false, false);     \
    ACC = __builtin_amdgcn_wmma_f32_16x16x32_bf16(false, (A_), false, (BL_),        \
                                                  (short)0, ACC, false, false);     \
    ACC = __builtin_amdgcn_wmma_f32_16x16x32_bf16(false, (AL_), false, (BH_),       \
                                                  (short)0, ACC, false, false);

// -------------------------------------------------------------------------
// Kernel 2: WMMA GEMM (bf16x3 ~= fp32) + fused clamp / rowmax / argmax /
// novelty. Block = 256 threads = 8 waves, owns 32 embedding rows.
// Each wave computes a 2M x 2N register tile block (4 accumulators):
//   rows: both 16-row groups; cols: ncg=wave (0..7), tiles at +0 / +16.
// Block covers 256 columns per outer iteration -> 16 iterations over C.
// A hi/lo staged to LDS once per block (ISA 16-bit A layout = 2 ds_b128
// per fragment); B fragments loaded directly from global (two 16-byte
// chunks per lane of one centroid row = exact ISA 32x16 bf16 B layout).
// 12 independent WMMAs per k-step hide the WMMA hazard windows.
// -------------------------------------------------------------------------
__global__ __launch_bounds__(256) void sim_gemm_kernel(
    const unsigned short* __restrict__ ehi, const unsigned short* __restrict__ elo,
    const unsigned short* __restrict__ chi, const unsigned short* __restrict__ clo,
    float* __restrict__ novelty, int* __restrict__ closest) {

    __shared__ __align__(16) unsigned short Ah[M_BLK * LDSS];
    __shared__ __align__(16) unsigned short Al[M_BLK * LDSS];
    __shared__ float redM[M_BLK][8];
    __shared__ int   redA[M_BLK][8];

    const int tid   = threadIdx.x;
    const int tbase = blockIdx.x * M_BLK;

    // ---- stage A (hi & lo) rows into LDS with padded stride ----
    for (int idx = tid; idx < M_BLK * (Dn / 8); idx += 256) {
        const int r = idx >> 7;       // Dn/8 = 128 16-byte chunks per row
        const int c = idx & 127;
        const uint4 vh = ((const uint4*)(ehi + (size_t)(tbase + r) * Dn))[c];
        const uint4 vl = ((const uint4*)(elo + (size_t)(tbase + r) * Dn))[c];
        *(uint4*)(&Ah[r * LDSS + c * 8]) = vh;
        *(uint4*)(&Al[r * LDSS + c * 8]) = vl;
    }
    __syncthreads();

    const int wave  = tid >> 5;
    const int lane  = tid & 31;
    const int ncg   = wave;        // 0..7 column group
    const int lrow  = lane & 15;
    const int lhalf = lane >> 4;

    // A-fragment LDS element offsets (k added in loop), one per 16-row group.
    // lanes 0-15: row=lrow, K = k0..k0+7 (q0) / k0+16..k0+23 (q1)
    // lanes 16-31: same row, +8 elements  (ISA 16-bit A 16x32 layout)
    const int aoff0 = lrow * LDSS + lhalf * 8;
    const int aoff1 = aoff0 + 16 * LDSS;

    float rmax0[8], rmax1[8];
    int   rarg0[8], rarg1[8];
    #pragma unroll
    for (int j = 0; j < 8; ++j) {
        rmax0[j] = -1.0f; rarg0[j] = 0;
        rmax1[j] = -1.0f; rarg1[j] = 0;
    }

    for (int nbase = 0; nbase < Cn; nbase += 256) {
        const int col0 = nbase + ncg * 32 + lrow;
        const int col1 = col0 + 16;
        // B layout: lane n (0-15): col n, K=k0..k0+7 / k0+16..k0+23;
        //           lane n+16: same col, K=k0+8..k0+15 / k0+24..k0+31.
        const unsigned short* pbh0 = chi + (size_t)col0 * Dn + lhalf * 8;
        const unsigned short* pbl0 = clo + (size_t)col0 * Dn + lhalf * 8;
        const unsigned short* pbh1 = chi + (size_t)col1 * Dn + lhalf * 8;
        const unsigned short* pbl1 = clo + (size_t)col1 * Dn + lhalf * 8;

        v8f acc00 = {};   // mgroup 0 x col tile 0
        v8f acc01 = {};   // mgroup 0 x col tile 1
        v8f acc10 = {};   // mgroup 1 x col tile 0
        v8f acc11 = {};   // mgroup 1 x col tile 1

        #pragma unroll 2
        for (int k0 = 0; k0 < Dn; k0 += 32) {
            FragU ah0, al0, ah1, al1, bh0, bl0, bh1, bl1;
            ah0.q[0] = *(const uint4*)(&Ah[aoff0 + k0]);
            ah0.q[1] = *(const uint4*)(&Ah[aoff0 + k0 + 16]);
            al0.q[0] = *(const uint4*)(&Al[aoff0 + k0]);
            al0.q[1] = *(const uint4*)(&Al[aoff0 + k0 + 16]);
            ah1.q[0] = *(const uint4*)(&Ah[aoff1 + k0]);
            ah1.q[1] = *(const uint4*)(&Ah[aoff1 + k0 + 16]);
            al1.q[0] = *(const uint4*)(&Al[aoff1 + k0]);
            al1.q[1] = *(const uint4*)(&Al[aoff1 + k0 + 16]);
            bh0.q[0] = *(const uint4*)(pbh0 + k0);
            bh0.q[1] = *(const uint4*)(pbh0 + k0 + 16);
            bl0.q[0] = *(const uint4*)(pbl0 + k0);
            bl0.q[1] = *(const uint4*)(pbl0 + k0 + 16);
            bh1.q[0] = *(const uint4*)(pbh1 + k0);
            bh1.q[1] = *(const uint4*)(pbh1 + k0 + 16);
            bl1.q[0] = *(const uint4*)(pbl1 + k0);
            bl1.q[1] = *(const uint4*)(pbl1 + k0 + 16);

            WMMA3(acc00, ah0.v, al0.v, bh0.v, bl0.v)
            WMMA3(acc10, ah1.v, al1.v, bh0.v, bl0.v)
            WMMA3(acc01, ah0.v, al0.v, bh1.v, bl1.v)
            WMMA3(acc11, ah1.v, al1.v, bh1.v, bl1.v)
        }

        // clamp to [0,1] and update running max/argmax.
        // C/D layout: element j -> row M = j + 8*lhalf, col N = lrow (+tile base).
        // col0 < col1 and nbase ascends -> strict '>' keeps FIRST max index.
        #pragma unroll
        for (int j = 0; j < 8; ++j) {
            float v;
            v = fminf(fmaxf(acc00[j], 0.0f), 1.0f);
            if (v > rmax0[j]) { rmax0[j] = v; rarg0[j] = col0; }
            v = fminf(fmaxf(acc01[j], 0.0f), 1.0f);
            if (v > rmax0[j]) { rmax0[j] = v; rarg0[j] = col1; }
            v = fminf(fmaxf(acc10[j], 0.0f), 1.0f);
            if (v > rmax1[j]) { rmax1[j] = v; rarg1[j] = col0; }
            v = fminf(fmaxf(acc11[j], 0.0f), 1.0f);
            if (v > rmax1[j]) { rmax1[j] = v; rarg1[j] = col1; }
        }
    }

    // ---- reduce across the 16 lanes of each half (columns of one row) ----
    #pragma unroll
    for (int mg = 0; mg < 2; ++mg) {
        #pragma unroll
        for (int j = 0; j < 8; ++j) {
            float m = mg ? rmax1[j] : rmax0[j];
            int   a = mg ? rarg1[j] : rarg0[j];
            #pragma unroll
            for (int msk = 8; msk >= 1; msk >>= 1) {
                const float om = __shfl_xor(m, msk, 32);
                const int   oa = __shfl_xor(a, msk, 32);
                if (om > m || (om == m && oa < a)) { m = om; a = oa; }
            }
            if (lrow == 0) {
                const int rr = mg * 16 + lhalf * 8 + j;   // row within block
                redM[rr][ncg] = m;
                redA[rr][ncg] = a;
            }
        }
    }
    __syncthreads();

    // ---- combine the 8 column groups, compute novelty, write out ----
    if (tid < M_BLK) {
        float m = redM[tid][0];
        int   a = redA[tid][0];
        #pragma unroll
        for (int g = 1; g < 8; ++g) {
            const float om = redM[tid][g];
            const int   oa = redA[tid][g];
            if (om > m || (om == m && oa < a)) { m = om; a = oa; }
        }
        float nv = sqrtf(fmaxf(1.0f - m, 0.0f));
        nv = fminf(fmaxf(nv, 0.0f), 1.0f);
        novelty[tbase + tid] = nv;
        closest[tbase + tid] = a;
    }
}

// -------------------------------------------------------------------------
// Launch: kernel1 (normalize + bf16 hi/lo split into ws), kernel2 (WMMA
// GEMM + fused novelty/argmax).
// d_ws layout (48 MB): Ehi | Elo | Chi | Clo  (bf16 as ushort)
// d_out: [0, Bn) float novelty, [Bn, 2*Bn) int32 closest.
// -------------------------------------------------------------------------
extern "C" void kernel_launch(void* const* d_in, const int* in_sizes, int n_in,
                              void* d_out, int out_size, void* d_ws, size_t ws_size,
                              hipStream_t stream) {
    const float* emb = (const float*)d_in[0];   // [Bn, Dn] fp32
    const float* cen = (const float*)d_in[1];   // [Cn, Dn] fp32

    unsigned short* ehi = (unsigned short*)d_ws;
    unsigned short* elo = ehi + (size_t)Bn * Dn;
    unsigned short* chi = elo + (size_t)Bn * Dn;
    unsigned short* clo = chi + (size_t)Cn * Dn;

    float* novelty = (float*)d_out;
    int*   closest = (int*)d_out + Bn;

    norm_split_kernel<<<Bn + Cn, 256, 0, stream>>>(emb, cen, ehi, elo, chi, clo);
    sim_gemm_kernel<<<Bn / M_BLK, 256, 0, stream>>>(ehi, elo, chi, clo, novelty, closest);
}
